// Attention_85564338471023
// MI455X (gfx1250) — matrix operand order repaired and checked by reading, not compiled, not run on hardware
//
#include <hip/hip_runtime.h>
#include <hip/hip_bf16.h>

typedef _Float16 v16h __attribute__((ext_vector_type(16)));
typedef _Float16 v8h  __attribute__((ext_vector_type(8)));
typedef float    v8f  __attribute__((ext_vector_type(8)));

#define BATCH 16
#define NTOK  1024
#define CCH   512
#define NH    8
#define DH    64
#define QKVC  1536
#define MTOT  (BATCH * NTOK)      // 16384
#define SCALE_F 0.125f            // 64^-0.5

#define WMMA_F16(a, b, c) \
    __builtin_amdgcn_wmma_f32_16x16x32_f16(false, (a), false, (b), (short)0, (c), false, false)

__device__ __forceinline__ v8f zero8() {
    v8f r = {0.f, 0.f, 0.f, 0.f, 0.f, 0.f, 0.f, 0.f};
    return r;
}

// Build a v16h A/B fragment from two contiguous 8-half (16B) runs.
__device__ __forceinline__ v16h ld16(const _Float16* p0, const _Float16* p1) {
    v8h a = *(const v8h*)p0;
    v8h b = *(const v8h*)p1;
    v16h r;
#pragma unroll
    for (int i = 0; i < 8; i++) { r[i] = a[i]; r[i + 8] = b[i]; }
    return r;
}

// ---------------------------------------------------------------------------
// Kernel 0: pack x and weights to f16 (weights transposed to [out_col][k]);
// fold 1/sqrt(d) into the Q columns of w_qkv / b_qkv.
// ---------------------------------------------------------------------------
__global__ void pack_inputs(const float* __restrict__ x, const float* __restrict__ w_qkv,
                            const float* __restrict__ b_qkv, const float* __restrict__ w_out,
                            _Float16* __restrict__ xh, _Float16* __restrict__ wqkvT,
                            _Float16* __restrict__ woutT, float* __restrict__ bq) {
    int stride = gridDim.x * blockDim.x;
    int tid0 = blockIdx.x * blockDim.x + threadIdx.x;
    // x: f32 -> f16, same layout, vectorized 4-wide
    for (int idx = tid0; idx < (MTOT * CCH) / 4; idx += stride) {
        float4 f = ((const float4*)x)[idx];
        v8h* dst = (v8h*)xh;   // write 4 halves via half of a v8h? use scalar pack:
        _Float16* p = xh + (size_t)idx * 4;
        p[0] = (_Float16)f.x; p[1] = (_Float16)f.y;
        p[2] = (_Float16)f.z; p[3] = (_Float16)f.w;
        (void)dst;
    }
    for (int idx = tid0; idx < QKVC * CCH; idx += stride) {
        int j = idx / CCH, k = idx - j * CCH;
        float w = w_qkv[(size_t)k * QKVC + j];
        if ((j % 192) < 64) w *= SCALE_F;          // Q columns pre-scaled
        wqkvT[idx] = (_Float16)w;
    }
    for (int idx = tid0; idx < CCH * CCH; idx += stride) {
        int co = idx / CCH, k = idx - co * CCH;
        woutT[idx] = (_Float16)w_out[(size_t)k * CCH + co];
    }
    for (int idx = tid0; idx < QKVC; idx += stride) {
        float bb = b_qkv[idx];
        if ((idx % 192) < 64) bb *= SCALE_F;
        bq[idx] = bb;
    }
}

// ---------------------------------------------------------------------------
// Kernel 1: qkv = xh @ WqkvT + b  (M=16384, K=512, N=1536).
// Each wave computes a 16x64 tile (A fragment reused by 4 WMMAs per k-step).
// Scatters results per-head into Q/K/V f16 buffers [b*NH+h][tok][d].
// ---------------------------------------------------------------------------
__global__ void qkv_gemm(const _Float16* __restrict__ xh, const _Float16* __restrict__ wT,
                         const float* __restrict__ bias,
                         _Float16* __restrict__ Q, _Float16* __restrict__ K,
                         _Float16* __restrict__ V) {
    int lane = threadIdx.x & 31;
    int wave = threadIdx.x >> 5;
    int gw = blockIdx.x * 8 + wave;
    const int NQ = QKVC / 64;                  // 24 column quads
    int tM = gw / NQ;
    int tN = gw - tM * NQ;
    int m0 = tM * 16, n0 = tN * 64;
    int lg = lane >> 4;                        // lane group (0/1)
    int mA = lane & 15;
    int nB = lane & 15;

    const _Float16* arow = xh + (size_t)(m0 + mA) * CCH;

    v8f acc0 = zero8(), acc1 = zero8(), acc2 = zero8(), acc3 = zero8();
#pragma unroll 4
    for (int k0 = 0; k0 < CCH; k0 += 32) {
        v16h a = ld16(arow + k0 + lg * 16, arow + k0 + lg * 16 + 8);
        v16h b;
        b = *(const v16h*)(wT + (size_t)(n0 + nB) * CCH + k0 + lg * 16);
        acc0 = WMMA_F16(a, b, acc0);
        b = *(const v16h*)(wT + (size_t)(n0 + 16 + nB) * CCH + k0 + lg * 16);
        acc1 = WMMA_F16(a, b, acc1);
        b = *(const v16h*)(wT + (size_t)(n0 + 32 + nB) * CCH + k0 + lg * 16);
        acc2 = WMMA_F16(a, b, acc2);
        b = *(const v16h*)(wT + (size_t)(n0 + 48 + nB) * CCH + k0 + lg * 16);
        acc3 = WMMA_F16(a, b, acc3);
    }

    // The whole 64-wide group [n0, n0+64) lies in one (head, q/k/v) segment.
    v8f accs[4] = {acc0, acc1, acc2, acc3};
#pragma unroll
    for (int t = 0; t < 4; t++) {
        int ncol = n0 + t * 16 + nB;
        float bcol = bias[ncol];
        int h = ncol / 192;
        int rr = ncol - h * 192;
        int which = rr >> 6;
        int d = rr & 63;
        _Float16* dst = (which == 0) ? Q : (which == 1) ? K : V;
#pragma unroll
        for (int r8 = 0; r8 < 8; r8++) {
            int bn = m0 + r8 + 8 * lg;         // D layout: row = reg + 8*(lane/16)
            int bb = bn >> 10, tok = bn & 1023;
            dst[(((size_t)(bb * NH + h)) * NTOK + tok) * DH + d] =
                (_Float16)(accs[t][r8] + bcol);
        }
    }
}

// ---------------------------------------------------------------------------
// Kernel 2: flash attention. Block = 8 waves = 128 query rows of one (b,h).
// Online softmax over key chunks of 32; K/V chunks staged in LDS; P transposed
// D-layout -> A-layout through per-wave LDS with s_wait_dscnt.
// ---------------------------------------------------------------------------
__global__ void attn_kernel(const _Float16* __restrict__ Q, const _Float16* __restrict__ K,
                            const _Float16* __restrict__ V, _Float16* __restrict__ O) {
    __shared__ __align__(32) _Float16 Kl[32 * 64];       // [y][d]
    __shared__ __align__(32) _Float16 Vt[64 * 32];       // [d][y] (transposed)
    __shared__ __align__(32) _Float16 Pl[8][16 * 32];    // per-wave P tile [m][y]

    int tid = threadIdx.x;
    int lane = tid & 31;
    int wave = tid >> 5;
    int bh = blockIdx.x >> 3;
    int qblk = blockIdx.x & 7;
    int q0 = qblk * 128 + wave * 16;
    int lg = lane >> 4;
    int mA = lane & 15;
    int nB = lane & 15;

    const _Float16* qrow = Q + ((size_t)bh * NTOK + q0 + mA) * DH;
    v16h aq0 = ld16(qrow + lg * 16,      qrow + lg * 16 + 8);   // d = 0..31
    v16h aq1 = ld16(qrow + 32 + lg * 16, qrow + 40 + lg * 16);   // d = 32..63

    v8f o0 = zero8(), o1 = zero8(), o2 = zero8(), o3 = zero8();
    float mr[8], lr[8];
#pragma unroll
    for (int i = 0; i < 8; i++) { mr[i] = -3.0e38f; lr[i] = 0.f; }

    const _Float16* Kb = K + (size_t)bh * NTOK * DH;
    const _Float16* Vb = V + (size_t)bh * NTOK * DH;

    for (int y0 = 0; y0 < NTOK; y0 += 32) {
        __syncthreads();   // previous chunk's LDS reads done before overwrite
        {
            int y = tid >> 3;
            int c = (tid & 7) * 8;
            // K chunk row-major
            *(v8h*)(Kl + y * 64 + c) = *(const v8h*)(Kb + (size_t)(y0 + y) * DH + c);
            // V chunk transposed [d][y]
            v8h vv = *(const v8h*)(Vb + (size_t)(y0 + y) * DH + c);
#pragma unroll
            for (int i = 0; i < 8; i++) Vt[(c + i) * 32 + y] = vv[i];
        }
        __syncthreads();

        // S = Q @ K^T for 32 keys: two 16x16 tiles, each accumulating d=64.
        v8f s0 = zero8(), s1 = zero8();
        {
            v16h bk;
            bk = *(const v16h*)(Kl + nB * 64 + lg * 16);             s0 = WMMA_F16(aq0, bk, s0);
            bk = *(const v16h*)(Kl + nB * 64 + 32 + lg * 16);        s0 = WMMA_F16(aq1, bk, s0);
            bk = *(const v16h*)(Kl + (16 + nB) * 64 + lg * 16);      s1 = WMMA_F16(aq0, bk, s1);
            bk = *(const v16h*)(Kl + (16 + nB) * 64 + 32 + lg * 16); s1 = WMMA_F16(aq1, bk, s1);
        }

        // Online softmax update (row = r8 + 8*lg, spread across 16 lanes).
        _Float16* Pw = Pl[wave];
#pragma unroll
        for (int r8 = 0; r8 < 8; r8++) {
            float cm = fmaxf(s0[r8], s1[r8]);
            cm = fmaxf(cm, __shfl_xor(cm, 1));
            cm = fmaxf(cm, __shfl_xor(cm, 2));
            cm = fmaxf(cm, __shfl_xor(cm, 4));
            cm = fmaxf(cm, __shfl_xor(cm, 8));
            float mnew  = fmaxf(mr[r8], cm);
            float alpha = __expf(mr[r8] - mnew);
            float p0 = __expf(s0[r8] - mnew);
            float p1 = __expf(s1[r8] - mnew);
            float rs = p0 + p1;
            rs += __shfl_xor(rs, 1);
            rs += __shfl_xor(rs, 2);
            rs += __shfl_xor(rs, 4);
            rs += __shfl_xor(rs, 8);
            lr[r8] = lr[r8] * alpha + rs;
            mr[r8] = mnew;
            o0[r8] *= alpha; o1[r8] *= alpha; o2[r8] *= alpha; o3[r8] *= alpha;
            int m = r8 + 8 * lg;
            Pw[m * 32 + nB]      = (_Float16)p0;
            Pw[m * 32 + 16 + nB] = (_Float16)p1;
        }
        // Cross-lane LDS RAW within the wave: wait out the DS stores.
        asm volatile("s_wait_dscnt 0" ::: "memory");

        // O += P @ V  (K-dim = 32 keys, N = 64 split into 4 tiles of 16)
        v16h pa = ld16(Pw + mA * 32 + lg * 16, Pw + mA * 32 + lg * 16 + 8);
        v16h bv;
        bv = *(const v16h*)(Vt + nB * 32 + lg * 16);        o0 = WMMA_F16(pa, bv, o0);
        bv = *(const v16h*)(Vt + (16 + nB) * 32 + lg * 16); o1 = WMMA_F16(pa, bv, o1);
        bv = *(const v16h*)(Vt + (32 + nB) * 32 + lg * 16); o2 = WMMA_F16(pa, bv, o2);
        bv = *(const v16h*)(Vt + (48 + nB) * 32 + lg * 16); o3 = WMMA_F16(pa, bv, o3);
    }

    // Normalize and scatter into [bn][h*64 + d] f16 for the output GEMM.
    int bb = bh >> 3, h = bh & 7;
#pragma unroll
    for (int r8 = 0; r8 < 8; r8++) {
        float inv = 1.0f / lr[r8];
        int tok = q0 + r8 + 8 * lg;
        size_t rowbase = ((size_t)(bb * NTOK + tok)) * CCH + h * DH;
        O[rowbase + nB]      = (_Float16)(o0[r8] * inv);
        O[rowbase + 16 + nB] = (_Float16)(o1[r8] * inv);
        O[rowbase + 32 + nB] = (_Float16)(o2[r8] * inv);
        O[rowbase + 48 + nB] = (_Float16)(o3[r8] * inv);
    }
}

// ---------------------------------------------------------------------------
// Kernel 3: out = attn @ WoutT + b_out + x   (M=16384, K=512, N=512), f32 out.
// Each wave computes a 16x64 tile.
// ---------------------------------------------------------------------------
__global__ void out_gemm(const _Float16* __restrict__ A, const _Float16* __restrict__ wT,
                         const float* __restrict__ bias, const float* __restrict__ x,
                         float* __restrict__ out) {
    int lane = threadIdx.x & 31;
    int wave = threadIdx.x >> 5;
    int gw = blockIdx.x * 8 + wave;
    const int NQ = CCH / 64;                   // 8 column quads
    int tM = gw / NQ;
    int tN = gw - tM * NQ;
    int m0 = tM * 16, n0 = tN * 64;
    int lg = lane >> 4;
    int mA = lane & 15;
    int nB = lane & 15;

    const _Float16* arow = A + (size_t)(m0 + mA) * CCH;

    v8f acc0 = zero8(), acc1 = zero8(), acc2 = zero8(), acc3 = zero8();
#pragma unroll 4
    for (int k0 = 0; k0 < CCH; k0 += 32) {
        v16h a = ld16(arow + k0 + lg * 16, arow + k0 + lg * 16 + 8);
        v16h b;
        b = *(const v16h*)(wT + (size_t)(n0 + nB) * CCH + k0 + lg * 16);
        acc0 = WMMA_F16(a, b, acc0);
        b = *(const v16h*)(wT + (size_t)(n0 + 16 + nB) * CCH + k0 + lg * 16);
        acc1 = WMMA_F16(a, b, acc1);
        b = *(const v16h*)(wT + (size_t)(n0 + 32 + nB) * CCH + k0 + lg * 16);
        acc2 = WMMA_F16(a, b, acc2);
        b = *(const v16h*)(wT + (size_t)(n0 + 48 + nB) * CCH + k0 + lg * 16);
        acc3 = WMMA_F16(a, b, acc3);
    }

    v8f accs[4] = {acc0, acc1, acc2, acc3};
#pragma unroll
    for (int t = 0; t < 4; t++) {
        int ncol = n0 + t * 16 + nB;
        float bcol = bias[ncol];
#pragma unroll
        for (int r8 = 0; r8 < 8; r8++) {
            size_t row = (size_t)(m0 + r8 + 8 * lg);
            out[row * CCH + ncol] = accs[t][r8] + bcol + x[row * CCH + ncol];
        }
    }
}

// ---------------------------------------------------------------------------
extern "C" void kernel_launch(void* const* d_in, const int* in_sizes, int n_in,
                              void* d_out, int out_size, void* d_ws, size_t ws_size,
                              hipStream_t stream) {
    const float* ft    = (const float*)d_in[0];   // (16,32,32,512) == (16384,512)
    const float* w_qkv = (const float*)d_in[1];   // (512,1536)
    const float* b_qkv = (const float*)d_in[2];   // (1536,)
    const float* w_out = (const float*)d_in[3];   // (512,512)
    const float* b_out = (const float*)d_in[4];   // (512,)
    float* outp = (float*)d_out;

    char* ws = (char*)d_ws;
    const size_t szQKV = (size_t)BATCH * NH * NTOK * DH * sizeof(_Float16); // 16 MB each
    _Float16* Qf   = (_Float16*)(ws);
    _Float16* Kf   = (_Float16*)(ws + szQKV);
    _Float16* Vf   = (_Float16*)(ws + 2 * szQKV);
    _Float16* Attn = (_Float16*)(ws + 3 * szQKV);                           // 16 MB
    _Float16* Xh   = (_Float16*)(ws + 4 * szQKV);                           // 16 MB
    char* wsw = ws + 5 * szQKV;
    _Float16* wqkvT = (_Float16*)(wsw);                                     // 1536*512 f16
    _Float16* woutT = (_Float16*)(wsw + (size_t)QKVC * CCH * sizeof(_Float16));
    float*    bq    = (float*)(wsw + (size_t)QKVC * CCH * sizeof(_Float16)
                                   + (size_t)CCH * CCH * sizeof(_Float16));

    pack_inputs<<<1024, 256, 0, stream>>>(ft, w_qkv, b_qkv, w_out, Xh, wqkvT, woutT, bq);

    // (16384/16) * (1536/64) = 24576 wave tiles / 8 waves per block
    qkv_gemm<<<3072, 256, 0, stream>>>(Xh, wqkvT, bq, Qf, Kf, Vf);

    // 128 (b,h) pairs * 8 query blocks of 128 rows
    attn_kernel<<<1024, 256, 0, stream>>>(Qf, Kf, Vf, Attn);

    // (16384/16) * (512/64) = 8192 wave tiles / 8
    out_gemm<<<1024, 256, 0, stream>>>(Attn, woutT, b_out, ft, outp);
}
